// LoraQKV_17970143167160
// MI455X (gfx1250) — compile-verified, matrix operand order checked by reading, not run on hardware
//
#include <hip/hip_runtime.h>
#include <hip/hip_bf16.h>
#include <math.h>

// ---------------------------------------------------------------------------
// MLA forward for MI455X (gfx1250, wave32).  All GEMM-shaped work is routed
// through v_wmma_f32_16x16x32_f16 (f16 in, f32 accumulate).  Staging is
// split load-phase / store-phase so global_load_b128 pipelines under the
// WMMA stream instead of serializing on s_wait_loadcnt.
// ---------------------------------------------------------------------------

typedef __attribute__((ext_vector_type(16))) _Float16 v16h;
typedef __attribute__((ext_vector_type(8)))  _Float16 v8h;
typedef __attribute__((ext_vector_type(4)))  _Float16 v4h;
typedef __attribute__((ext_vector_type(8)))  float    v8f;

#define NHEADS   32
#define HDIM     64
#define RDIM     64
#define SEQ      2048
#define HID      2048
#define QRANK    1024
#define KVRANK   896
#define QK_SCALE 0.125f     // 64^-0.5
#define RMS_EPS  1e-6f

// ---------------------------------------------------------------------------
// WMMA helpers
// ---------------------------------------------------------------------------
__device__ __forceinline__ v8f wmma_f16(v16h a, v16h b, v8f c) {
  return __builtin_amdgcn_wmma_f32_16x16x32_f16(false, a, false, b, (short)0, c,
                                                false, false);
}

__device__ __forceinline__ v8f zero8() {
  v8f z = {0.f, 0.f, 0.f, 0.f, 0.f, 0.f, 0.f, 0.f};
  return z;
}

// CDNA5 16-bit A/B fragment: lanes 0-15 hold K {0..7,16..23}; lanes 16-31
// hold K {8..15,24..31} (ISA 7.12.2).  Two ds_load_b128 + register concat.
__device__ __forceinline__ v16h frag_ld(const _Float16* p, int grp) {
  v8h lo = *(const v8h*)(p + 8 * grp);
  v8h hi = *(const v8h*)(p + 16 + 8 * grp);
  return __builtin_shufflevector(lo, hi, 0, 1, 2, 3, 4, 5, 6, 7,
                                 8, 9, 10, 11, 12, 13, 14, 15);
}

__device__ __forceinline__ v4h cvt4(float4 v) {
  v4h h = {(_Float16)v.x, (_Float16)v.y, (_Float16)v.z, (_Float16)v.w};
  return h;
}

// ===========================================================================
// GEMM, 128x128 tile (N % 128 == 0):  C[M,N] = A[M,K] * W[N,K]^T
// 256 threads / 8 waves; wave (wm=wave&3, wn=wave>>2) computes 32x64.
// Double-buffered LDS; loads for step s+1 issue before step-s WMMAs, the
// cvt+ds_store lands after them.
// ===========================================================================
__device__ __forceinline__ void g128_ld(const float* __restrict__ A,
                                        const float* __restrict__ W,
                                        int mbase, int nbase, int k0, int K,
                                        int tid, float4* ra, float4* rb) {
#pragma unroll
  for (int j = 0; j < 4; ++j) {               // A,B: 128x32 each
    int g  = j * 256 + tid;
    int r  = g >> 3;
    int c4 = (g & 7) * 4;
    ra[j] = *(const float4*)&A[(size_t)(mbase + r) * K + k0 + c4];
    rb[j] = *(const float4*)&W[(size_t)(nbase + r) * K + k0 + c4];
  }
}

__device__ __forceinline__ void g128_st(_Float16* __restrict__ dA,
                                        _Float16* __restrict__ dB, int tid,
                                        const float4* ra, const float4* rb) {
#pragma unroll
  for (int j = 0; j < 4; ++j) {
    int g  = j * 256 + tid;
    int r  = g >> 3;
    int c4 = (g & 7) * 4;
    *(v4h*)&dA[r * 32 + c4] = cvt4(ra[j]);
    *(v4h*)&dB[r * 32 + c4] = cvt4(rb[j]);
  }
}

__global__ __launch_bounds__(256) void gemm_wmma128_kernel(
    const float* __restrict__ A, const float* __restrict__ W,
    float* __restrict__ C, int M, int N, int K) {
  __shared__ alignas(16) _Float16 lA[2][128 * 32];   // 2 x 8 KB
  __shared__ alignas(16) _Float16 lB[2][128 * 32];   // 2 x 8 KB

  const int tid  = threadIdx.x;
  const int wave = tid >> 5;
  const int lane = tid & 31;
  const int grp  = lane >> 4;
  const int l15  = lane & 15;
  const int wm   = wave & 3;    // 0..3 -> 32-row strip
  const int wn   = wave >> 2;   // 0..1 -> 64-col strip
  const int mbase = blockIdx.y * 128;
  const int nbase = blockIdx.x * 128;

  v8f acc0[4], acc1[4];
#pragma unroll
  for (int j = 0; j < 4; ++j) { acc0[j] = zero8(); acc1[j] = zero8(); }

  float4 ra[4], rb[4];
  const int nsteps = K >> 5;
  g128_ld(A, W, mbase, nbase, 0, K, tid, ra, rb);
  g128_st(lA[0], lB[0], tid, ra, rb);

  for (int s = 0; s < nsteps; ++s) {
    __syncthreads();
    if (s + 1 < nsteps)                      // issue next-step global loads
      g128_ld(A, W, mbase, nbase, (s + 1) << 5, K, tid, ra, rb);

    const _Float16* pa = lA[s & 1];
    const _Float16* pb = lB[s & 1];
    v16h a0 = frag_ld(pa + (wm * 32 + l15) * 32, grp);
    v16h a1 = frag_ld(pa + (wm * 32 + 16 + l15) * 32, grp);
#pragma unroll
    for (int jn = 0; jn < 4; ++jn) {
      v16h b = frag_ld(pb + (wn * 64 + jn * 16 + l15) * 32, grp);
      acc0[jn] = wmma_f16(a0, b, acc0[jn]);
      acc1[jn] = wmma_f16(a1, b, acc1[jn]);
    }

    if (s + 1 < nsteps)                      // wait + convert lands here
      g128_st(lA[(s + 1) & 1], lB[(s + 1) & 1], tid, ra, rb);
  }

#pragma unroll
  for (int jn = 0; jn < 4; ++jn)
#pragma unroll
    for (int r = 0; r < 8; ++r) {
      int m0 = mbase + wm * 32 + r + 8 * grp;  // C/D: M = vgpr + 8*grp
      int n0 = nbase + wn * 64 + jn * 16 + l15;
      C[(size_t)m0 * N + n0]        = acc0[jn][r];
      C[(size_t)(m0 + 16) * N + n0] = acc1[jn][r];
    }
}

// ===========================================================================
// GEMM, 128x64 tile (N % 64 == 0, used for N=960):  same pipeline,
// wave (wm=wave&3, wn=wave>>2) computes 32x32.
// ===========================================================================
__device__ __forceinline__ void g64_ld(const float* __restrict__ A,
                                       const float* __restrict__ W,
                                       int mbase, int nbase, int k0, int K,
                                       int tid, float4* ra, float4* rb) {
#pragma unroll
  for (int j = 0; j < 4; ++j) {               // A: 128x32
    int g  = j * 256 + tid;
    int r  = g >> 3;
    int c4 = (g & 7) * 4;
    ra[j] = *(const float4*)&A[(size_t)(mbase + r) * K + k0 + c4];
  }
#pragma unroll
  for (int j = 0; j < 2; ++j) {               // B: 64x32
    int g  = j * 256 + tid;
    int r  = g >> 3;
    int c4 = (g & 7) * 4;
    rb[j] = *(const float4*)&W[(size_t)(nbase + r) * K + k0 + c4];
  }
}

__device__ __forceinline__ void g64_st(_Float16* __restrict__ dA,
                                       _Float16* __restrict__ dB, int tid,
                                       const float4* ra, const float4* rb) {
#pragma unroll
  for (int j = 0; j < 4; ++j) {
    int g  = j * 256 + tid;
    int r  = g >> 3;
    int c4 = (g & 7) * 4;
    *(v4h*)&dA[r * 32 + c4] = cvt4(ra[j]);
  }
#pragma unroll
  for (int j = 0; j < 2; ++j) {
    int g  = j * 256 + tid;
    int r  = g >> 3;
    int c4 = (g & 7) * 4;
    *(v4h*)&dB[r * 32 + c4] = cvt4(rb[j]);
  }
}

__global__ __launch_bounds__(256) void gemm_wmma64_kernel(
    const float* __restrict__ A, const float* __restrict__ W,
    float* __restrict__ C, int M, int N, int K) {
  __shared__ alignas(16) _Float16 lA[2][128 * 32];
  __shared__ alignas(16) _Float16 lB[2][64 * 32];

  const int tid  = threadIdx.x;
  const int wave = tid >> 5;
  const int lane = tid & 31;
  const int grp  = lane >> 4;
  const int l15  = lane & 15;
  const int wm   = wave & 3;
  const int wn   = wave >> 2;
  const int mbase = blockIdx.y * 128;
  const int nbase = blockIdx.x * 64;

  v8f c00 = zero8(), c01 = zero8(), c10 = zero8(), c11 = zero8();

  float4 ra[4], rb[2];
  const int nsteps = K >> 5;
  g64_ld(A, W, mbase, nbase, 0, K, tid, ra, rb);
  g64_st(lA[0], lB[0], tid, ra, rb);

  for (int s = 0; s < nsteps; ++s) {
    __syncthreads();
    if (s + 1 < nsteps)
      g64_ld(A, W, mbase, nbase, (s + 1) << 5, K, tid, ra, rb);

    const _Float16* pa = lA[s & 1];
    const _Float16* pb = lB[s & 1];
    v16h a0 = frag_ld(pa + (wm * 32 + l15) * 32, grp);
    v16h a1 = frag_ld(pa + (wm * 32 + 16 + l15) * 32, grp);
    v16h b0 = frag_ld(pb + (wn * 32 + l15) * 32, grp);
    v16h b1 = frag_ld(pb + (wn * 32 + 16 + l15) * 32, grp);

    c00 = wmma_f16(a0, b0, c00);
    c01 = wmma_f16(a0, b1, c01);
    c10 = wmma_f16(a1, b0, c10);
    c11 = wmma_f16(a1, b1, c11);

    if (s + 1 < nsteps)
      g64_st(lA[(s + 1) & 1], lB[(s + 1) & 1], tid, ra, rb);
  }

#pragma unroll
  for (int r = 0; r < 8; ++r) {
    int m0 = mbase + wm * 32 + r + 8 * grp;
    int m1 = m0 + 16;
    int n0 = nbase + wn * 32 + l15;
    C[(size_t)m0 * N + n0]      = c00[r];
    C[(size_t)m0 * N + n0 + 16] = c01[r];
    C[(size_t)m1 * N + n0]      = c10[r];
    C[(size_t)m1 * N + n0 + 16] = c11[r];
  }
}

// ---------------------------------------------------------------------------
// RMSNorm over rows:  out = in * rsqrt(mean(in^2)+eps) * w
// ---------------------------------------------------------------------------
__global__ __launch_bounds__(256) void rmsnorm_kernel(
    const float* __restrict__ in, float* __restrict__ out,
    const float* __restrict__ w, int width, int in_stride, int out_stride) {
  __shared__ float red[256];
  const int row = blockIdx.x;
  const int tid = threadIdx.x;
  const float* ip = in + (size_t)row * in_stride;
  float ss = 0.f;
  for (int i = tid; i < width; i += 256) {
    float v = ip[i];
    ss += v * v;
  }
  red[tid] = ss;
  __syncthreads();
  for (int s = 128; s > 0; s >>= 1) {
    if (tid < s) red[tid] += red[tid + s];
    __syncthreads();
  }
  float scale = rsqrtf(red[0] / (float)width + RMS_EPS);
  float* op = out + (size_t)row * out_stride;
  for (int i = tid; i < width; i += 256) op[i] = ip[i] * scale * w[i];
}

// ---------------------------------------------------------------------------
// RoPE on q rope sections (in place).  grid (SEQ, 8), block 256.
// ---------------------------------------------------------------------------
__global__ __launch_bounds__(256) void rope_q_kernel(
    float* __restrict__ q, const float* __restrict__ cs,
    const float* __restrict__ sn) {
  const int s = blockIdx.x;
  const int t = blockIdx.y * 256 + threadIdx.x;
  const int h = t >> 6;
  const int i = t & 63;
  size_t base = (size_t)s * (NHEADS * 128) + h * 128 + 64;
  float x  = q[base + i];
  float xr = (i < 32) ? -q[base + i + 32] : q[base + i - 32];
  float v  = x * cs[s * RDIM + i] + xr * sn[s * RDIM + i];
  __syncthreads();
  q[base + i] = v;
}

__global__ __launch_bounds__(64) void rope_k_kernel(
    const float* __restrict__ ckv, const float* __restrict__ cs,
    const float* __restrict__ sn, float* __restrict__ krope) {
  const int s = blockIdx.x;
  const int i = threadIdx.x;
  const float* kp = ckv + (size_t)s * (KVRANK + RDIM) + KVRANK;
  float x  = kp[i];
  float xr = (i < 32) ? -kp[i + 32] : kp[i - 32];
  krope[(size_t)s * RDIM + i] = x * cs[s * RDIM + i] + xr * sn[s * RDIM + i];
}

// ---------------------------------------------------------------------------
// Causal flash attention, one head x 64 q-rows per block; 4 waves, each wave
// owns 16 q rows.  QK^T and P*V via WMMA; online softmax in registers.
// ---------------------------------------------------------------------------
__global__ __launch_bounds__(128) void mla_attn_kernel(
    const float* __restrict__ q, const float* __restrict__ kv,
    const float* __restrict__ krope, float* __restrict__ out) {
  __shared__ alignas(16) _Float16 sQ[64 * 128];       // 16 KB
  __shared__ alignas(16) _Float16 sK[64 * 128];       // 16 KB
  __shared__ alignas(16) _Float16 sVT[64 * 64];       //  8 KB
  __shared__ alignas(16) _Float16 sP[4 * 16 * 64];    //  8 KB

  const int h     = blockIdx.y;
  const int qb    = blockIdx.x;
  const int qbase = qb * 64;
  const int tid   = threadIdx.x;
  const int wave  = tid >> 5;
  const int lane  = tid & 31;
  const int grp   = lane >> 4;
  const int l15   = lane & 15;

  float4 buf[8];

  // ---- stage Q tile (scaled), batched 8 loads per wait ----
#pragma unroll
  for (int half = 0; half < 2; ++half) {
#pragma unroll
    for (int j = 0; j < 8; ++j) {
      int g  = (half * 8 + j) * 128 + tid;
      int r  = g >> 5;
      int c4 = (g & 31) * 4;
      buf[j] = *(const float4*)&q[(size_t)(qbase + r) * (NHEADS * 128) +
                                  h * 128 + c4];
    }
#pragma unroll
    for (int j = 0; j < 8; ++j) {
      int g  = (half * 8 + j) * 128 + tid;
      int r  = g >> 5;
      int c4 = (g & 31) * 4;
      v4h hv = {(_Float16)(buf[j].x * QK_SCALE), (_Float16)(buf[j].y * QK_SCALE),
                (_Float16)(buf[j].z * QK_SCALE), (_Float16)(buf[j].w * QK_SCALE)};
      *(v4h*)&sQ[r * 128 + c4] = hv;
    }
  }
  __syncthreads();

  v16h qf[4];
#pragma unroll
  for (int kk = 0; kk < 4; ++kk)
    qf[kk] = frag_ld(sQ + (wave * 16 + l15) * 128 + kk * 32, grp);

  float m_s[8], l_s[8];
  v8f O[4];
#pragma unroll
  for (int r = 0; r < 8; ++r) { m_s[r] = -1e30f; l_s[r] = 0.f; }
#pragma unroll
  for (int j = 0; j < 4; ++j) O[j] = zero8();

  int qrow[8];
#pragma unroll
  for (int r = 0; r < 8; ++r) qrow[r] = qbase + wave * 16 + r + 8 * grp;

  for (int kb = 0; kb <= qb; ++kb) {
    const int kbase = kb * 64;
    __syncthreads();
    // ---- stage K tile [64 keys][128 d], batched ----
#pragma unroll
    for (int half = 0; half < 2; ++half) {
#pragma unroll
      for (int j = 0; j < 8; ++j) {
        int g  = (half * 8 + j) * 128 + tid;
        int r  = g >> 5;
        int c4 = (g & 31) * 4;              // 4-group never straddles d=64
        const float* src =
            (c4 < 64) ? &kv[(size_t)(kbase + r) * (NHEADS * 128) + h * 128 + c4]
                      : &krope[(size_t)(kbase + r) * RDIM + (c4 - 64)];
        buf[j] = *(const float4*)src;
      }
#pragma unroll
      for (int j = 0; j < 8; ++j) {
        int g  = (half * 8 + j) * 128 + tid;
        int r  = g >> 5;
        int c4 = (g & 31) * 4;
        *(v4h*)&sK[r * 128 + c4] = cvt4(buf[j]);
      }
    }
    // ---- stage V^T tile [64 d][64 keys], batched loads ----
#pragma unroll
    for (int j = 0; j < 8; ++j) {
      int g   = j * 128 + tid;
      int key = g & 63;
      int d4  = (g >> 6) * 4;
      buf[j] = *(const float4*)&kv[(size_t)(kbase + key) * (NHEADS * 128) +
                                   h * 128 + 64 + d4];
    }
#pragma unroll
    for (int j = 0; j < 8; ++j) {
      int g   = j * 128 + tid;
      int key = g & 63;
      int d4  = (g >> 6) * 4;
      sVT[(d4 + 0) * 64 + key] = (_Float16)buf[j].x;
      sVT[(d4 + 1) * 64 + key] = (_Float16)buf[j].y;
      sVT[(d4 + 2) * 64 + key] = (_Float16)buf[j].z;
      sVT[(d4 + 3) * 64 + key] = (_Float16)buf[j].w;
    }
    // ---- prefetch next K/V tile (global_prefetch_b8) ----
    if (kb < qb) {
      const float* pk = &kv[(size_t)(kbase + 64 + (tid & 63)) * (NHEADS * 128) +
                            h * 128 + ((tid >> 6) << 6)];
      __builtin_prefetch(pk, 0, 1);
      __builtin_prefetch(pk + 32, 0, 1);
    }
    __syncthreads();

    // ---- S = Q * K^T ----
    v8f S[4];
#pragma unroll
    for (int n = 0; n < 4; ++n) {
      v8f acc = zero8();
#pragma unroll
      for (int kk = 0; kk < 4; ++kk) {
        v16h b = frag_ld(sK + (n * 16 + l15) * 128 + kk * 32, grp);
        acc = wmma_f16(qf[kk], b, acc);
      }
      S[n] = acc;
    }

    // ---- causal mask + online softmax ----
#pragma unroll
    for (int r = 0; r < 8; ++r) {
      float mx = m_s[r];
#pragma unroll
      for (int n = 0; n < 4; ++n) {
        int kglob = kbase + n * 16 + l15;
        float sv  = (kglob <= qrow[r]) ? S[n][r] : -1e30f;
        S[n][r]   = sv;
        mx = fmaxf(mx, sv);
      }
#pragma unroll
      for (int off = 1; off < 16; off <<= 1)
        mx = fmaxf(mx, __shfl_xor(mx, off, 32));
      float alpha = __expf(m_s[r] - mx);
      m_s[r] = mx;
      float rowsum = 0.f;
#pragma unroll
      for (int n = 0; n < 4; ++n) {
        float p = __expf(S[n][r] - mx);
        S[n][r] = p;
        rowsum += p;
      }
#pragma unroll
      for (int off = 1; off < 16; off <<= 1)
        rowsum += __shfl_xor(rowsum, off, 32);
      l_s[r] = l_s[r] * alpha + rowsum;
#pragma unroll
      for (int j = 0; j < 4; ++j) O[j] = O[j] * alpha;
    }

    // ---- P -> per-wave LDS -> A fragments ----
#pragma unroll
    for (int n = 0; n < 4; ++n)
#pragma unroll
      for (int r = 0; r < 8; ++r)
        sP[wave * 1024 + (r + 8 * grp) * 64 + n * 16 + l15] = (_Float16)S[n][r];

    v16h pa0 = frag_ld(sP + wave * 1024 + l15 * 64, grp);
    v16h pa1 = frag_ld(sP + wave * 1024 + l15 * 64 + 32, grp);

    // ---- O += P * V ----
#pragma unroll
    for (int j = 0; j < 4; ++j) {
      v16h vb0 = frag_ld(sVT + (j * 16 + l15) * 64, grp);
      v16h vb1 = frag_ld(sVT + (j * 16 + l15) * 64 + 32, grp);
      O[j] = wmma_f16(pa0, vb0, O[j]);
      O[j] = wmma_f16(pa1, vb1, O[j]);
    }
  }

  // ---- epilogue ----
#pragma unroll
  for (int j = 0; j < 4; ++j)
#pragma unroll
    for (int r = 0; r < 8; ++r) {
      int row = qbase + wave * 16 + r + 8 * grp;
      out[(size_t)row * (NHEADS * HDIM) + h * HDIM + j * 16 + l15] =
          O[j][r] / l_s[r];
    }
}

// ---------------------------------------------------------------------------
// Orchestration
// ---------------------------------------------------------------------------
extern "C" void kernel_launch(void* const* d_in, const int* in_sizes, int n_in,
                              void* d_out, int out_size, void* d_ws,
                              size_t ws_size, hipStream_t stream) {
  (void)in_sizes; (void)n_in; (void)out_size; (void)ws_size;

  const float* hidden   = (const float*)d_in[0];  // [S,H]
  const float* cosp     = (const float*)d_in[1];  // [S,64]
  const float* sinp     = (const float*)d_in[2];  // [S,64]
  const float* q_a_w    = (const float*)d_in[3];  // [1024,2048]
  const float* q_a_ln   = (const float*)d_in[4];  // [1024]
  const float* q_b_w    = (const float*)d_in[5];  // [4096,1024]
  const float* kv_a_w   = (const float*)d_in[6];  // [960,2048]
  const float* kv_a_ln  = (const float*)d_in[7];  // [896]
  const float* kv_b_w   = (const float*)d_in[8];  // [4096,896]
  const float* o_w      = (const float*)d_in[9];  // [2048,2048]
  float* outp           = (float*)d_out;          // [S,2048]

  float* ws      = (float*)d_ws;
  float* q_lat   = ws;                      ws += (size_t)SEQ * QRANK;
  float* ckv     = ws;                      ws += (size_t)SEQ * (KVRANK + RDIM);
  float* ckvn    = ws;                      ws += (size_t)SEQ * KVRANK;
  float* qbuf    = ws;                      ws += (size_t)SEQ * NHEADS * 128;
  float* kvbuf   = ws;                      ws += (size_t)SEQ * NHEADS * 128;
  float* kropeb  = ws;                      ws += (size_t)SEQ * RDIM;
  float* attn_o  = ws;                      ws += (size_t)SEQ * NHEADS * HDIM;

  const dim3 blk256(256);

  // 1) q_lat_raw = hidden @ q_a_w^T          [2048,1024]
  gemm_wmma128_kernel<<<dim3(QRANK / 128, SEQ / 128), blk256, 0, stream>>>(
      hidden, q_a_w, q_lat, SEQ, QRANK, HID);
  // 2) ckv = hidden @ kv_a_w^T               [2048,960]  (N % 128 != 0)
  gemm_wmma64_kernel<<<dim3((KVRANK + RDIM) / 64, SEQ / 128), blk256, 0,
                      stream>>>(hidden, kv_a_w, ckv, SEQ, KVRANK + RDIM, HID);
  // 3) rmsnorm q_lat (in place)
  rmsnorm_kernel<<<SEQ, 256, 0, stream>>>(q_lat, q_lat, q_a_ln, QRANK, QRANK,
                                          QRANK);
  // 4) rmsnorm c_kv -> ckvn
  rmsnorm_kernel<<<SEQ, 256, 0, stream>>>(ckv, ckvn, kv_a_ln, KVRANK,
                                          KVRANK + RDIM, KVRANK);
  // 5) q = q_lat @ q_b_w^T                   [2048,4096]
  gemm_wmma128_kernel<<<dim3(NHEADS * 128 / 128, SEQ / 128), blk256, 0,
                       stream>>>(q_lat, q_b_w, qbuf, SEQ, NHEADS * 128, QRANK);
  // 6) kv = ckvn @ kv_b_w^T                  [2048,4096]
  gemm_wmma128_kernel<<<dim3(NHEADS * 128 / 128, SEQ / 128), blk256, 0,
                       stream>>>(ckvn, kv_b_w, kvbuf, SEQ, NHEADS * 128,
                                 KVRANK);
  // 7) RoPE on q rope halves (in place)
  rope_q_kernel<<<dim3(SEQ, 8), 256, 0, stream>>>(qbuf, cosp, sinp);
  // 8) RoPE on shared k_rope -> kropeb
  rope_k_kernel<<<SEQ, 64, 0, stream>>>(ckv, cosp, sinp, kropeb);
  // 9) causal flash attention                [2048,2048]
  mla_attn_kernel<<<dim3(SEQ / 64, NHEADS), 128, 0, stream>>>(
      qbuf, kvbuf, kropeb, attn_o);
  // 10) out = attn_o @ o_w^T                 [2048,2048]
  gemm_wmma128_kernel<<<dim3(HID / 128, SEQ / 128), blk256, 0, stream>>>(
      attn_o, o_w, outp, SEQ, HID, HID);
}
